// SPGATLayer_10093173145803
// MI455X (gfx1250) — compile-verified, matrix operand order checked by reading, not run on hardware
//
#include <hip/hip_runtime.h>
#include <stdint.h>

typedef __bf16 bf16_t;
typedef __attribute__((ext_vector_type(16))) __bf16 v16bf;
typedef __attribute__((ext_vector_type(8)))  __bf16 v8bf;
typedef __attribute__((ext_vector_type(8)))  float  v8f;

#define DIM_IN  512
#define DIM_OUT 256

// ---------------- fp32 -> bf16 (round to nearest even), 4 elts/thread -------
__device__ __forceinline__ unsigned short f2bf(float f) {
  unsigned u = __float_as_uint(f);
  u += 0x7FFFu + ((u >> 16) & 1u);
  return (unsigned short)(u >> 16);
}

__global__ void cvt_f32_bf16x4(const float4* __restrict__ in,
                               ushort4* __restrict__ out, int n4) {
  int i = blockIdx.x * blockDim.x + threadIdx.x;
  if (i >= n4) return;
  float4 f = in[i];
  ushort4 o;
  o.x = f2bf(f.x); o.y = f2bf(f.y); o.z = f2bf(f.z); o.w = f2bf(f.w);
  out[i] = o;
}

__global__ void fill_f32x4(float4* __restrict__ p, float v, int n4) {
  int i = blockIdx.x * blockDim.x + threadIdx.x;
  if (i < n4) p[i] = make_float4(v, v, v, v);
}

// ---------------- bf16 WMMA GEMM: C[n x 256] = A[n x 512] * W[256 x 512]^T ---
// ISA 16-bit A/B fragment layout: lane L holds row(A)/col(B) = L%16,
// K-chunks at k0+8*(L/16) and k0+16+8*(L/16)  (two 16B loads).
__device__ __forceinline__ v16bf load_frag(const bf16_t* __restrict__ row,
                                           int k0, int half) {
  union { v16bf v; v8bf h[2]; } u;
  u.h[0] = *(const v8bf*)(row + k0 + half * 8);
  u.h[1] = *(const v8bf*)(row + k0 + 16 + half * 8);
  return u.v;
}

// Block: 256 threads = 8 waves; block tile = 64 rows x 256 cols.
// Wave w: cols [w*32, w*32+32) -> 2 B fragments, 4x2 accumulators.
__global__ void __launch_bounds__(256)
gemm_bf16_wmma(const bf16_t* __restrict__ A, const bf16_t* __restrict__ W,
               float* __restrict__ C, int n) {
  const int lane = threadIdx.x & 31;
  const int wave = threadIdx.x >> 5;
  const int half = lane >> 4;
  const int lm   = lane & 15;
  const int rowBase = blockIdx.x * 64;
  const int col     = wave * 32;

  const bf16_t* wrow0 = W + (size_t)(col + lm) * DIM_IN;        // B col tile 0
  const bf16_t* wrow1 = W + (size_t)(col + 16 + lm) * DIM_IN;   // B col tile 1
  const bf16_t* arow[4];
#pragma unroll
  for (int r = 0; r < 4; ++r) {
    int row = rowBase + r * 16 + lm;
    if (row >= n) row = n - 1;        // clamp OOB loads; stores predicated
    arow[r] = A + (size_t)row * DIM_IN;
  }

  v8f acc[4][2] = {};
  for (int k0 = 0; k0 < DIM_IN; k0 += 32) {
    v16bf b0 = load_frag(wrow0, k0, half);
    v16bf b1 = load_frag(wrow1, k0, half);
#pragma unroll
    for (int r = 0; r < 4; ++r) {
      v16bf a = load_frag(arow[r], k0, half);
      acc[r][0] = __builtin_amdgcn_wmma_f32_16x16x32_bf16(
          false, a, false, b0, (short)0, acc[r][0], false, false);
      acc[r][1] = __builtin_amdgcn_wmma_f32_16x16x32_bf16(
          false, a, false, b1, (short)0, acc[r][1], false, false);
    }
  }

  // C layout: VGPR v, lane L -> M = v + 8*(L/16), N = L%16
  if (rowBase + 64 <= n) {            // fast path: unpredicated stores
#pragma unroll
    for (int r = 0; r < 4; ++r) {
#pragma unroll
      for (int c = 0; c < 2; ++c) {
        float* cp = C + (size_t)(rowBase + r * 16 + half * 8) * DIM_OUT
                      + col + c * 16 + lm;
#pragma unroll
        for (int v = 0; v < 8; ++v)
          cp[(size_t)v * DIM_OUT] = acc[r][c][v];
      }
    }
  } else {                            // tail block only
#pragma unroll
    for (int r = 0; r < 4; ++r) {
#pragma unroll
      for (int c = 0; c < 2; ++c) {
        int row0 = rowBase + r * 16 + half * 8;
#pragma unroll
        for (int v = 0; v < 8; ++v) {
          int row = row0 + v;
          if (row < n) C[(size_t)row * DIM_OUT + col + c * 16 + lm] = acc[r][c][v];
        }
      }
    }
  }
}

// ---------------- s[row] = dot(z[row, :256], a[:256]) (wave per row) --------
__global__ void rowdot(const float* __restrict__ z, const float* __restrict__ a,
                       float* __restrict__ s, int n) {
  int row  = (blockIdx.x * blockDim.x + threadIdx.x) >> 5;
  int lane = threadIdx.x & 31;
  if (row >= n) return;
  const float* zr = z + (size_t)row * DIM_OUT;
  float p = 0.f;
#pragma unroll
  for (int i = 0; i < 8; ++i) p += zr[lane + 32 * i] * a[lane + 32 * i];
#pragma unroll
  for (int off = 16; off > 0; off >>= 1) p += __shfl_xor(p, off, 32);
  if (lane == 0) s[row] = p;
}

// ---------------- edge kernels ----------------
__device__ __forceinline__ void atomicMaxF(float* addr, float v) {
  if (v >= 0.f) atomicMax((int*)addr, __float_as_int(v));
  else          atomicMin((unsigned int*)addr, __float_as_uint(v));
}

__global__ void edge_logits(const float* __restrict__ ssrc,
                            const float* __restrict__ sdst,
                            const int* __restrict__ src, const int* __restrict__ dst,
                            float* __restrict__ e, float* __restrict__ m, int E) {
  int j = blockIdx.x * blockDim.x + threadIdx.x;
  if (j >= E) return;
  float v = ssrc[src[j]] + sdst[dst[j]];
  v = v > 0.f ? v : 0.01f * v;         // leaky_relu, slope 0.01
  e[j] = v;
  atomicMaxF(&m[dst[j]], v);
}

__global__ void edge_exp(const float* __restrict__ e, const float* __restrict__ m,
                         const int* __restrict__ dst, float* __restrict__ ex,
                         float* __restrict__ denom, int E) {
  int j = blockIdx.x * blockDim.x + threadIdx.x;
  if (j >= E) return;
  float v = __expf(e[j] - m[dst[j]]);
  ex[j] = v;
  atomicAdd(&denom[dst[j]], v);
}

__global__ void edge_aggr(const float* __restrict__ ex,
                          const float* __restrict__ denom,
                          const int* __restrict__ src, const int* __restrict__ dst,
                          const float* __restrict__ zs, float* __restrict__ out, int E) {
  int j    = (blockIdx.x * blockDim.x + threadIdx.x) >> 5;   // wave per edge
  int lane = threadIdx.x & 31;
  if (j >= E) return;
  int s = src[j], d = dst[j];
  float alpha = ex[j] / denom[d];
  const float* zr = zs + (size_t)s * DIM_OUT;
  float* o = out + (size_t)d * DIM_OUT;
#pragma unroll
  for (int i = 0; i < 8; ++i) {
    int c = lane + 32 * i;
    atomicAdd(o + c, alpha * zr[c]);   // z_s (51MB) resident in 192MB L2
  }
}

// ---------------- host orchestration ----------------
extern "C" void kernel_launch(void* const* d_in, const int* in_sizes, int n_in,
                              void* d_out, int out_size, void* d_ws, size_t ws_size,
                              hipStream_t stream) {
  const float* h_s  = (const float*)d_in[0];
  const float* h_p  = (const float*)d_in[1];
  const float* W    = (const float*)d_in[2];
  const float* attn = (const float*)d_in[3];
  const int*   src  = (const int*)d_in[4];
  const int*   dst  = (const int*)d_in[5];
  const int NS = in_sizes[0] / DIM_IN;
  const int NP = in_sizes[1] / DIM_IN;
  const int E  = in_sizes[4];
  float* out = (float*)d_out;

  char* ws = (char*)d_ws;
  size_t off = 0;
  auto alloc = [&](size_t bytes) -> void* {
    void* p = ws + off;
    off = (off + bytes + 255) & ~(size_t)255;
    return p;
  };
  unsigned short* hsb = (unsigned short*)alloc((size_t)NS * DIM_IN * 2);
  unsigned short* hpb = (unsigned short*)alloc((size_t)NP * DIM_IN * 2);
  unsigned short* Wb  = (unsigned short*)alloc((size_t)DIM_OUT * DIM_IN * 2);
  float* z_s   = (float*)alloc((size_t)NS * DIM_OUT * 4);
  float* z_p   = (float*)alloc((size_t)NP * DIM_OUT * 4);
  float* s_src = (float*)alloc((size_t)NS * 4);
  float* s_dst = (float*)alloc((size_t)NP * 4);
  float* eArr  = (float*)alloc((size_t)E * 4);
  float* exArr = (float*)alloc((size_t)E * 4);
  float* mArr  = (float*)alloc((size_t)NP * 4);
  float* dArr  = (float*)alloc((size_t)NP * 4);

  const int T = 256;
  // 1) bf16 copies for the WMMA path (all sizes are multiples of 4)
  cvt_f32_bf16x4<<<(NS * DIM_IN / 4 + T - 1) / T, T, 0, stream>>>(
      (const float4*)h_s, (ushort4*)hsb, NS * DIM_IN / 4);
  cvt_f32_bf16x4<<<(NP * DIM_IN / 4 + T - 1) / T, T, 0, stream>>>(
      (const float4*)h_p, (ushort4*)hpb, NP * DIM_IN / 4);
  cvt_f32_bf16x4<<<(DIM_OUT * DIM_IN / 4 + T - 1) / T, T, 0, stream>>>(
      (const float4*)W, (ushort4*)Wb, DIM_OUT * DIM_IN / 4);

  // 2) node projections via v_wmma_f32_16x16x32_bf16 (A read once)
  gemm_bf16_wmma<<<(NS + 63) / 64, T, 0, stream>>>(
      (const bf16_t*)hsb, (const bf16_t*)Wb, z_s, NS);
  gemm_bf16_wmma<<<(NP + 63) / 64, T, 0, stream>>>(
      (const bf16_t*)hpb, (const bf16_t*)Wb, z_p, NP);

  // 3) attention score halves (fp32)
  rowdot<<<(NS + 7) / 8, T, 0, stream>>>(z_s, attn, s_src, NS);
  rowdot<<<(NP + 7) / 8, T, 0, stream>>>(z_p, attn + DIM_OUT, s_dst, NP);

  // 4) init accumulators (harness poisons buffers each run)
  fill_f32x4<<<(NP / 4 + T - 1) / T, T, 0, stream>>>(
      (float4*)mArr, -__builtin_inff(), NP / 4);
  fill_f32x4<<<(NP / 4 + T - 1) / T, T, 0, stream>>>(
      (float4*)dArr, 0.f, NP / 4);
  fill_f32x4<<<(NP * DIM_OUT / 4 + T - 1) / T, T, 0, stream>>>(
      (float4*)out, 0.f, NP * DIM_OUT / 4);

  // 5) segment softmax + weighted aggregation
  edge_logits<<<(E + T - 1) / T, T, 0, stream>>>(s_src, s_dst, src, dst, eArr, mArr, E);
  edge_exp<<<(E + T - 1) / T, T, 0, stream>>>(eArr, mArr, dst, exArr, dArr, E);
  edge_aggr<<<(E + 7) / 8, T, 0, stream>>>(exArr, dArr, src, dst, z_s, out, E);
}
